// CudaGRUCell_4887672783102
// MI455X (gfx1250) — matrix operand order repaired and checked by reading, not compiled, not run on hardware
//
#include <hip/hip_runtime.h>
#include <hip/hip_bf16.h>
#include <math.h>

// ---------------------------------------------------------------------------
// GRU fused persistent scan for MI455X (gfx1250), wave32 + WMMA bf16.
//
// DIM=1024, B=32, T=1024.
//   z  = sigmoid(x_t@Wz^T + h@Uz^T + bz)     -> ONE fused WMMA accumulator
//   r  = sigmoid(x_t@Wr^T + h@Ur^T + br)     -> ONE fused WMMA accumulator
//   ht = tanh  (x_t@Wh^T + bh + r*(h@Uh^T))  -> xh and uh kept separate
//   h  = (1-z)*h + z*ht
//
// 64 column strips of 16; one wave per strip. 8 accumulator v8f sets
// (z/r/xh/uh x 2 M-tiles) = 64 VGPRs -> no spills in the serial inner loop.
// Weights pre-packed to B-operand tiles, k = 16*(lane/16) + j (12MB bf16, L2-resident).
// x_t / h_prev staged per step into LDS as A-operand tiles in the same k order.
// ---------------------------------------------------------------------------

typedef __bf16 v16bf __attribute__((ext_vector_type(16)));
typedef __bf16 v8bf  __attribute__((ext_vector_type(8)));
typedef float  v8f   __attribute__((ext_vector_type(8)));
typedef float  v4f   __attribute__((ext_vector_type(4)));

#define GDIM     1024
#define GBATCH   32
#define GT       1024
#define KITERS   32          // GDIM / 32 (K per WMMA)
#define STRIPS   64          // GDIM / 16 (N tiles)
#define NWG      8
#define WAVES    8
#define TILE_ELEMS 512       // one 16x32 (A) or 32x16 (B) bf16 tile
#define MAT_ELEMS  (STRIPS * KITERS * TILE_ELEMS)   // 1,048,576 bf16 per matrix

__device__ __forceinline__ float fast_sigmoid(float v) {
    // 1/(1+exp(-v)) with v_rcp_f32 instead of the IEEE divide expansion.
    return __builtin_amdgcn_rcpf(1.f + __expf(-v));
}
__device__ __forceinline__ float fast_tanh(float v) {
    // tanh(v) = 2*sigmoid(2v) - 1
    return 2.f * __builtin_amdgcn_rcpf(1.f + __expf(-2.f * v)) - 1.f;
}

// ---------------------------------------------------------------------------
// Pack the 6 weight matrices (Wz, Wr, Wh, Uz, Ur, Uh) from f32 row-major
// [n=out_col][k=in_dim] into bf16 B-operand tiles:
//   tile(strip s, kIter ki): 512 bf16, element (lane*16 + j) where
//     lane<16 : N=lane,    K=j
//     lane>=16: N=lane-16, K=16+j
// One thread packs 8 contiguous K (16 bytes).
// ---------------------------------------------------------------------------
__global__ void gru_pack_weights(const float* __restrict__ W_zr,
                                 const float* __restrict__ U_zr,
                                 const float* __restrict__ W_h,
                                 const float* __restrict__ U_h,
                                 __bf16* __restrict__ packed) {
    int c = blockIdx.x * blockDim.x + threadIdx.x;   // 6*64*32*32*2 = 786432
    int half = c & 1;
    int lane = (c >> 1) & 31;
    int ki   = (c >> 6) & 31;
    int s    = (c >> 11) & 63;
    int mat  = c >> 17;                               // 0..5
    int n = s * 16 + (lane & 15);                     // output column e
    int k = ki * 32 + ((lane < 16) ? half * 8 : 16 + half * 8);

    const float* src;
    switch (mat) {
        case 0:  src = W_zr + (size_t)n * GDIM;            break; // Wz
        case 1:  src = W_zr + (size_t)(GDIM + n) * GDIM;   break; // Wr
        case 2:  src = W_h  + (size_t)n * GDIM;            break; // Wh
        case 3:  src = U_zr + (size_t)n * GDIM;            break; // Uz
        case 4:  src = U_zr + (size_t)(GDIM + n) * GDIM;   break; // Ur
        default: src = U_h  + (size_t)n * GDIM;            break; // Uh
    }
    const float* p = src + k;
    v4f a = *(const v4f*)p;
    v4f b = *(const v4f*)(p + 4);
    v8bf v;
    v[0]=(__bf16)a[0]; v[1]=(__bf16)a[1]; v[2]=(__bf16)a[2]; v[3]=(__bf16)a[3];
    v[4]=(__bf16)b[0]; v[5]=(__bf16)b[1]; v[6]=(__bf16)b[2]; v[7]=(__bf16)b[3];
    size_t off = (size_t)mat * MAT_ELEMS
               + (size_t)(s * KITERS + ki) * TILE_ELEMS
               + lane * 16 + half * 8;
    *(v8bf*)(packed + off) = v;
}

// ---------------------------------------------------------------------------
// Persistent fused GRU scan. grid = 8 WGs x 256 threads (8 waves of 32).
// Wave w of WG g owns strip s = g*8+w (output columns [16s, 16s+16)).
// ---------------------------------------------------------------------------
__launch_bounds__(256, 1)
__global__ void gru_scan(const float* __restrict__ x,     // [T][B][DIM] f32
                         const float* __restrict__ b_zr,  // [2*DIM]
                         const float* __restrict__ b_h,   // [DIM]
                         const __bf16* __restrict__ packed,
                         float* __restrict__ hbuf0,       // [B][DIM] ping
                         float* __restrict__ hbuf1,       // [B][DIM] pong
                         float* __restrict__ out,         // [T][B][DIM]
                         unsigned* __restrict__ bar) {
    // A-operand tiles, k = 16*(lane/16) + j (same as B), tile idx = mt*KITERS + ki.
    __shared__ v16bf ldsX[2 * KITERS * 32];   // 64 KB
    __shared__ v16bf ldsH[2 * KITERS * 32];   // 64 KB

    const int tid   = threadIdx.x;
    const int lane  = tid & 31;
    const int wv    = tid >> 5;
    const int strip = blockIdx.x * WAVES + wv;
    const int ecol  = strip * 16 + (lane & 15);   // this lane's output column

    const float bz = b_zr[ecol];
    const float br = b_zr[GDIM + ecol];
    const float bh = b_h[ecol];

    // Per-matrix base of this strip's packed B tiles (lane-offset folded in).
    const size_t stripOff = (size_t)strip * KITERS * TILE_ELEMS + (size_t)lane * 16;
    const __bf16* pWz = packed + 0ull * MAT_ELEMS + stripOff;
    const __bf16* pWr = packed + 1ull * MAT_ELEMS + stripOff;
    const __bf16* pWh = packed + 2ull * MAT_ELEMS + stripOff;
    const __bf16* pUz = packed + 3ull * MAT_ELEMS + stripOff;
    const __bf16* pUr = packed + 4ull * MAT_ELEMS + stripOff;
    const __bf16* pUh = packed + 5ull * MAT_ELEMS + stripOff;

    for (int t = 0; t < GT; ++t) {
        // Ternary (not pointer array) keeps these in the GLOBAL address space
        // so h loads/stores lower to global_*, not flat_* (no DScnt coupling).
        const float* hprev = (t & 1) ? hbuf1 : hbuf0;
        float*       hnext = (t & 1) ? hbuf0 : hbuf1;
        const float* xt = x + (size_t)t * (GBATCH * GDIM);

        // ---- stage x_t and h_prev into LDS A-tile layout (f32 -> bf16) ----
        __bf16* lx = (__bf16*)ldsX;
        __bf16* lh = (__bf16*)ldsH;
        for (int c = tid; c < 4096; c += 256) {           // 16-byte chunks
            int half = c & 1;
            int ln   = (c >> 1) & 31;
            int ki   = (c >> 6) & 31;
            int mt   = (c >> 11) & 1;
            int m      = ln & 15;
            int kstart = (ln < 16) ? (half ? 8 : 0) : (half ? 24 : 16);
            int b = mt * 16 + m;
            int d = ki * 32 + kstart;
            const float* ph = hprev + (size_t)b * GDIM + d;
            const float* px = xt    + (size_t)b * GDIM + d;
            v4f h0v = *(const v4f*)ph;  v4f h1v = *(const v4f*)(ph + 4);
            v4f x0v = *(const v4f*)px;  v4f x1v = *(const v4f*)(px + 4);
            v8bf hb, xb;
#pragma unroll
            for (int i = 0; i < 4; ++i) {
                hb[i] = (__bf16)h0v[i]; hb[4 + i] = (__bf16)h1v[i];
                xb[i] = (__bf16)x0v[i]; xb[4 + i] = (__bf16)x1v[i];
            }
            int loff = (mt * KITERS + ki) * TILE_ELEMS + ln * 16 + half * 8;
            *(v8bf*)(lh + loff) = hb;
            *(v8bf*)(lx + loff) = xb;
            if (t + 1 < GT)  // warm L2 for next step's x tile (global_prefetch_b8)
                __builtin_prefetch(px + GBATCH * GDIM, 0, 0);
        }
        __syncthreads();

        // ---- fused GEMMs: 8 live accumulators, 12 WMMAs per K-iter --------
        v8f accZ0  = (v8f){0,0,0,0,0,0,0,0}, accZ1  = (v8f){0,0,0,0,0,0,0,0};
        v8f accR0  = (v8f){0,0,0,0,0,0,0,0}, accR1  = (v8f){0,0,0,0,0,0,0,0};
        v8f accXH0 = (v8f){0,0,0,0,0,0,0,0}, accXH1 = (v8f){0,0,0,0,0,0,0,0};
        v8f accUH0 = (v8f){0,0,0,0,0,0,0,0}, accUH1 = (v8f){0,0,0,0,0,0,0,0};

#pragma unroll 1
        for (int ki = 0; ki < KITERS; ++ki) {
            v16bf ax0 = ldsX[ki * 32 + lane];
            v16bf ax1 = ldsX[(KITERS + ki) * 32 + lane];
            v16bf ah0 = ldsH[ki * 32 + lane];
            v16bf ah1 = ldsH[(KITERS + ki) * 32 + lane];
            const size_t ko = (size_t)ki * TILE_ELEMS;

            v16bf bm;
            // z gate: x@Wz^T + h@Uz^T fused into one accumulator pair
            bm = *(const v16bf*)(pWz + ko);
            accZ0 = __builtin_amdgcn_wmma_f32_16x16x32_bf16(false, ax0, false, bm, (short)0, accZ0, false, false);
            accZ1 = __builtin_amdgcn_wmma_f32_16x16x32_bf16(false, ax1, false, bm, (short)0, accZ1, false, false);
            bm = *(const v16bf*)(pUz + ko);
            accZ0 = __builtin_amdgcn_wmma_f32_16x16x32_bf16(false, ah0, false, bm, (short)0, accZ0, false, false);
            accZ1 = __builtin_amdgcn_wmma_f32_16x16x32_bf16(false, ah1, false, bm, (short)0, accZ1, false, false);
            // r gate: x@Wr^T + h@Ur^T fused
            bm = *(const v16bf*)(pWr + ko);
            accR0 = __builtin_amdgcn_wmma_f32_16x16x32_bf16(false, ax0, false, bm, (short)0, accR0, false, false);
            accR1 = __builtin_amdgcn_wmma_f32_16x16x32_bf16(false, ax1, false, bm, (short)0, accR1, false, false);
            bm = *(const v16bf*)(pUr + ko);
            accR0 = __builtin_amdgcn_wmma_f32_16x16x32_bf16(false, ah0, false, bm, (short)0, accR0, false, false);
            accR1 = __builtin_amdgcn_wmma_f32_16x16x32_bf16(false, ah1, false, bm, (short)0, accR1, false, false);
            // candidate: xh and uh kept separate (r scales only uh)
            bm = *(const v16bf*)(pWh + ko);
            accXH0 = __builtin_amdgcn_wmma_f32_16x16x32_bf16(false, ax0, false, bm, (short)0, accXH0, false, false);
            accXH1 = __builtin_amdgcn_wmma_f32_16x16x32_bf16(false, ax1, false, bm, (short)0, accXH1, false, false);
            bm = *(const v16bf*)(pUh + ko);
            accUH0 = __builtin_amdgcn_wmma_f32_16x16x32_bf16(false, ah0, false, bm, (short)0, accUH0, false, false);
            accUH1 = __builtin_amdgcn_wmma_f32_16x16x32_bf16(false, ah1, false, bm, (short)0, accUH1, false, false);
        }

        // ---- gating + update. D layout: lane holds column N=lane%16,
        //      VGPR g holds row M=g (lanes 0-15) or 8+g (lanes 16-31). -------
#pragma unroll
        for (int mt = 0; mt < 2; ++mt) {
            const v8f az  = mt ? accZ1  : accZ0;
            const v8f ar  = mt ? accR1  : accR0;
            const v8f axh = mt ? accXH1 : accXH0;
            const v8f auh = mt ? accUH1 : accUH0;
#pragma unroll
            for (int g = 0; g < 8; ++g) {
                int b = mt * 16 + ((lane & 16) ? (8 + g) : g);
                float z  = fast_sigmoid(az[g] + bz);
                float r  = fast_sigmoid(ar[g] + br);
                float hp = hprev[(size_t)b * GDIM + ecol];
                float ht = fast_tanh(axh[g] + bh + r * auh[g]);
                float hn = (1.f - z) * hp + z * ht;
                out[(size_t)t * (GBATCH * GDIM) + (size_t)b * GDIM + ecol] = hn;
                hnext[(size_t)b * GDIM + ecol] = hn;
            }
        }

        // ---- device-wide barrier (monotone counter, no reset race) --------
        __threadfence();
        __syncthreads();
        if (tid == 0) {
            atomicAdd(bar, 1u);
            unsigned target = (unsigned)(t + 1) * NWG;
            while (__hip_atomic_load(bar, __ATOMIC_ACQUIRE,
                                     __HIP_MEMORY_SCOPE_AGENT) < target) {
                __builtin_amdgcn_s_sleep(2);
            }
        }
        __syncthreads();
        __threadfence();
    }
}

// ---------------------------------------------------------------------------
extern "C" void kernel_launch(void* const* d_in, const int* in_sizes, int n_in,
                              void* d_out, int out_size, void* d_ws, size_t ws_size,
                              hipStream_t stream) {
    const float* x    = (const float*)d_in[0];
    const float* h0   = (const float*)d_in[1];
    const float* W_zr = (const float*)d_in[2];
    const float* U_zr = (const float*)d_in[3];
    const float* W_h  = (const float*)d_in[4];
    const float* U_h  = (const float*)d_in[5];
    const float* b_zr = (const float*)d_in[6];
    const float* b_h  = (const float*)d_in[7];
    float* out = (float*)d_out;

    char* ws = (char*)d_ws;
    unsigned* bar  = (unsigned*)ws;                               // 256 B
    float* hbuf0   = (float*)(ws + 256);                          // 128 KB
    float* hbuf1   = hbuf0 + GBATCH * GDIM;                       // 128 KB
    __bf16* packed = (__bf16*)(ws + 256 + 2 * GBATCH * GDIM * sizeof(float)); // 12 MB

    hipMemsetAsync(bar, 0, 256, stream);
    hipMemcpyAsync(hbuf0, h0, GBATCH * GDIM * sizeof(float),
                   hipMemcpyDeviceToDevice, stream);

    // 6 matrices * 64 strips * 32 kIters * 64 chunks = 786432 threads
    gru_pack_weights<<<786432 / 256, 256, 0, stream>>>(W_zr, U_zr, W_h, U_h, packed);

    gru_scan<<<NWG, 256, 0, stream>>>(x, b_zr, b_h, packed,
                                      hbuf0, hbuf1, out, bar);
}
